// CustomPatchEmbedding_7009386627450
// MI455X (gfx1250) — matrix operand order repaired and checked by reading, not compiled, not run on hardware
//
#include <hip/hip_runtime.h>

// Problem constants (fixed by the reference)
#define BATCH   32
#define CH      3
#define IMG_H   512
#define IMG_W   512
#define NPATCH  196
#define PH      16
#define PW      16
#define KDIM    768            // CH*PH*PW
#define DDIM    768
#define MTOT    (BATCH * NPATCH)   // 6272 = 49 * 128

typedef __bf16 v16bf __attribute__((ext_vector_type(16)));
typedef __bf16 v8bf  __attribute__((ext_vector_type(8)));
typedef float  v8f   __attribute__((ext_vector_type(8)));
typedef int    v4i   __attribute__((ext_vector_type(4)));

// ---- bit-exact RNE f32 -> bf16 helpers (no __bf16 arithmetic needed) ----
__device__ __forceinline__ unsigned short f32_to_bf16(float f) {
    union { float f; unsigned int u; } c; c.f = f;
    const unsigned int u = c.u;
    const unsigned int r = u + 0x7FFFu + ((u >> 16) & 1u);  // round-to-nearest-even
    return (unsigned short)(r >> 16);
}
__device__ __forceinline__ float bf16_to_f32(unsigned short h) {
    union { unsigned int u; float f; } c; c.u = ((unsigned int)h) << 16;
    return c.f;
}
__device__ __forceinline__ void split_bf16(float v, unsigned short& hi, unsigned short& lo) {
    hi = f32_to_bf16(v);
    lo = f32_to_bf16(v - bf16_to_f32(hi));
}

// ---- async copy: 16 bytes global -> LDS, tracked by ASYNCcnt ----
__device__ __forceinline__ void async_cp16(const unsigned short* g, unsigned short* l) {
    __builtin_amdgcn_global_load_async_to_lds_b128(
        (__attribute__((address_space(1))) v4i*)g,
        (__attribute__((address_space(3))) v4i*)l,
        /*offset=*/0, /*cpol=*/0);
}
__device__ __forceinline__ void wait_async0() {
    asm volatile("s_wait_asynccnt 0x0" ::: "memory");
}

// ---------------------------------------------------------------------------
// Kernel 1: gather clamped 3x16x16 patches -> bf16 hi/lo planes Ah/Al [M][K].
// ---------------------------------------------------------------------------
__global__ __launch_bounds__(256)
void gather_patches_kernel(const float* __restrict__ x,
                           const int* __restrict__ centers,
                           unsigned short* __restrict__ Ah,
                           unsigned short* __restrict__ Al) {
    const int p = blockIdx.x;              // patch 0..6271
    const int b = p / NPATCH;
    const int n = p - b * NPATCH;

    const int h = centers[(b * NPATCH + n) * 2 + 0];
    const int w = centers[(b * NPATCH + n) * 2 + 1];
    int sh = h - PH / 2; sh = sh < 0 ? 0 : (sh > IMG_H - PH ? IMG_H - PH : sh);
    int sw = w - PW / 2; sw = sw < 0 ? 0 : (sw > IMG_W - PW ? IMG_W - PW : sw);

    const float* __restrict__ img = x + (size_t)b * CH * IMG_H * IMG_W;
    unsigned short* __restrict__ dh = Ah + (size_t)p * KDIM;
    unsigned short* __restrict__ dl = Al + (size_t)p * KDIM;

    #pragma unroll
    for (int i = 0; i < 3; ++i) {
        const int k   = threadIdx.x + i * 256;   // 0..767
        const int c   = k >> 8;
        const int rem = k & 255;
        const int r   = rem >> 4;
        const int col = rem & 15;
        const float v = img[((size_t)c * IMG_H + (sh + r)) * IMG_W + (sw + col)];
        unsigned short hi, lo; split_bf16(v, hi, lo);
        dh[k] = hi; dl[k] = lo;
    }
}

// ---------------------------------------------------------------------------
// Kernel 2: W[K][D] f32 -> transposed bf16 planes Wt_hi/Wt_lo [D][K].
// ---------------------------------------------------------------------------
__global__ __launch_bounds__(256)
void prep_weights_kernel(const float* __restrict__ Wm,
                         unsigned short* __restrict__ Wth,
                         unsigned short* __restrict__ Wtl) {
    const int idx = blockIdx.x * 256 + threadIdx.x;   // 0 .. 768*768-1
    const int k = idx / DDIM;
    const int d = idx - k * DDIM;
    unsigned short hi, lo; split_bf16(Wm[idx], hi, lo);
    Wth[(size_t)d * KDIM + k] = hi;
    Wtl[(size_t)d * KDIM + k] = lo;
}

// ---------------------------------------------------------------------------
// Kernel 3: out = A*W + b via split-bf16 WMMA (Ah*Bh + Ah*Bl + Al*Bh, f32 acc).
// Block tile 128x128, 8 waves (2x4), wave tile 64x32 = 4x2 WMMA tiles, BK=32.
// Double-buffered LDS, async global->LDS staging overlapped with WMMA.
// ---------------------------------------------------------------------------
#define BM     128
#define BN     128
#define BK     32
#define LDT    40     // padded LDS stride in bf16 elements (80B rows, 16B-aligned)
#define NCHUNK (KDIM / BK)   // 24

__global__ __launch_bounds__(256)
void patch_gemm_wmma_bf16x3_kernel(const unsigned short* __restrict__ Ah,
                                   const unsigned short* __restrict__ Al,
                                   const unsigned short* __restrict__ Bh,  // [D][K]
                                   const unsigned short* __restrict__ Bl,  // [D][K]
                                   const float* __restrict__ bias,
                                   float* __restrict__ out) {
    // Double-buffered planes: 2 * 4 * 128*40*2B = 80 KB (WGP has 320 KB).
    __shared__ unsigned short AhL[2][BM * LDT];
    __shared__ unsigned short AlL[2][BM * LDT];
    __shared__ unsigned short BhL[2][BN * LDT];
    __shared__ unsigned short BlL[2][BN * LDT];

    const int tid  = threadIdx.x;
    const int lane = tid & 31;
    const int wave = tid >> 5;        // 0..7
    const int wm   = wave >> 2;       // 0..1 -> 64-row slab
    const int wn   = wave & 3;        // 0..3 -> 32-col slab
    const int half = lane >> 4;       // K-half selector in fragments
    const int l16  = lane & 15;       // A row / B,C column

    const int m0 = blockIdx.x * BM;   // 49 tiles
    const int n0 = blockIdx.y * BN;   // 6 tiles

    // Per-thread staging assignment: 2 x 16B per plane per chunk.
    const int row0 = tid >> 2;              // rows tid/4 and tid/4 + 64
    const int q0   = (tid & 3) * 8;         // bf16 offset within 32-wide row

    auto stage_chunk = [&](int kc, int buf) {
        #pragma unroll
        for (int i = 0; i < 2; ++i) {
            const int row = row0 + i * 64;
            const size_t ga = (size_t)(m0 + row) * KDIM + kc + q0;
            const size_t gb = (size_t)(n0 + row) * KDIM + kc + q0;
            const int lo = row * LDT + q0;
            async_cp16(Ah + ga, &AhL[buf][lo]);
            async_cp16(Al + ga, &AlL[buf][lo]);
            async_cp16(Bh + gb, &BhL[buf][lo]);
            async_cp16(Bl + gb, &BlL[buf][lo]);
        }
    };

    v8f acc[4][2];
    #pragma unroll
    for (int i = 0; i < 4; ++i)
        #pragma unroll
        for (int j = 0; j < 2; ++j)
            acc[i][j] = 0.0f;

    stage_chunk(0, 0);   // prologue: chunk 0 -> buffer 0

    for (int c = 0; c < NCHUNK; ++c) {
        const int cur = c & 1;
        wait_async0();        // chunk c resident in LDS (this wave's copies)
        __syncthreads();      // ... and everyone else's; prev buffer now free

        if (c + 1 < NCHUNK)
            stage_chunk((c + 1) * BK, cur ^ 1);   // overlap with WMMA below

        // B fragments: 32x16, elems 0-7 = K 8*half+0..7, elems 8-15 =
        // K 16+8*half+0..7, column = lane%16.  LDS B is column-major.
        v16bf bh[2], bl[2];
        #pragma unroll
        for (int j = 0; j < 2; ++j) {
            const int cidx = wn * 32 + j * 16 + l16;
            const int base = cidx * LDT + 8 * half;
            const v8bf bh0 = *(const v8bf*)(&BhL[cur][base]);
            const v8bf bh1 = *(const v8bf*)(&BhL[cur][base + 16]);
            const v8bf bl0 = *(const v8bf*)(&BlL[cur][base]);
            const v8bf bl1 = *(const v8bf*)(&BlL[cur][base + 16]);
            bh[j] = __builtin_shufflevector(bh0, bh1, 0,1,2,3,4,5,6,7,8,9,10,11,12,13,14,15);
            bl[j] = __builtin_shufflevector(bl0, bl1, 0,1,2,3,4,5,6,7,8,9,10,11,12,13,14,15);
        }

        #pragma unroll
        for (int i = 0; i < 4; ++i) {
            // A fragments: 16x32 16-bit layout: elems 0-7 = K 8*half+0..7,
            // elems 8-15 = K 16+8*half+0..7, row = lane%16.
            const int abase = (wm * 64 + i * 16 + l16) * LDT;
            const v8bf ah0 = *(const v8bf*)(&AhL[cur][abase + 8 * half]);
            const v8bf ah1 = *(const v8bf*)(&AhL[cur][abase + 16 + 8 * half]);
            const v8bf al0 = *(const v8bf*)(&AlL[cur][abase + 8 * half]);
            const v8bf al1 = *(const v8bf*)(&AlL[cur][abase + 16 + 8 * half]);
            const v16bf afh = __builtin_shufflevector(ah0, ah1, 0,1,2,3,4,5,6,7,8,9,10,11,12,13,14,15);
            const v16bf afl = __builtin_shufflevector(al0, al1, 0,1,2,3,4,5,6,7,8,9,10,11,12,13,14,15);
            #pragma unroll
            for (int j = 0; j < 2; ++j) {
                // low-order correction terms first, then the dominant product
                acc[i][j] = __builtin_amdgcn_wmma_f32_16x16x32_bf16(
                    false, afl, false, bh[j], (short)0, acc[i][j], false, false);
                acc[i][j] = __builtin_amdgcn_wmma_f32_16x16x32_bf16(
                    false, afh, false, bl[j], (short)0, acc[i][j], false, false);
                acc[i][j] = __builtin_amdgcn_wmma_f32_16x16x32_bf16(
                    false, afh, false, bh[j], (short)0, acc[i][j], false, false);
            }
        }
    }

    // Epilogue: bias add + store (C/D layout: VGPR v holds rows v (lanes 0-15)
    // and v+8 (lanes 16-31), col = lane%16).
    #pragma unroll
    for (int j = 0; j < 2; ++j) {
        const int col = n0 + wn * 32 + j * 16 + l16;
        const float bv = bias[col];
        #pragma unroll
        for (int i = 0; i < 4; ++i) {
            const int rbase = m0 + wm * 64 + i * 16 + half * 8;
            #pragma unroll
            for (int v = 0; v < 8; ++v) {
                out[(size_t)(rbase + v) * DDIM + col] = acc[i][j][v] + bv;
            }
        }
    }
}

// ---------------------------------------------------------------------------
extern "C" void kernel_launch(void* const* d_in, const int* in_sizes, int n_in,
                              void* d_out, int out_size, void* d_ws, size_t ws_size,
                              hipStream_t stream) {
    const float* x       = (const float*)d_in[0];   // [32,3,512,512] f32
    const int*   centers = (const int*)d_in[1];     // [32,196,2] i32
    const float* Wm      = (const float*)d_in[2];   // [768,768] f32
    const float* bias    = (const float*)d_in[3];   // [768] f32
    float*       out     = (float*)d_out;           // [32,196,768] f32

    // Workspace layout (bf16 planes): Ah, Al [M][K]; Wt_hi, Wt_lo [D][K]
    unsigned short* ws  = (unsigned short*)d_ws;
    unsigned short* Ah  = ws;
    unsigned short* Al  = Ah + (size_t)MTOT * KDIM;
    unsigned short* Wth = Al + (size_t)MTOT * KDIM;
    unsigned short* Wtl = Wth + (size_t)DDIM * KDIM;

    prep_weights_kernel<<<(KDIM * DDIM) / 256, 256, 0, stream>>>(Wm, Wth, Wtl);
    gather_patches_kernel<<<MTOT, 256, 0, stream>>>(x, centers, Ah, Al);

    dim3 grid(MTOT / BM, DDIM / BN);                // 49 x 6
    patch_gemm_wmma_bf16x3_kernel<<<grid, 256, 0, stream>>>(Ah, Al, Wth, Wtl, bias, out);
}
